// GCN_68994354643303
// MI455X (gfx1250) — matrix operand order repaired and checked by reading, not compiled, not run on hardware
//
#include <hip/hip_runtime.h>

typedef __attribute__((ext_vector_type(16))) _Float16 v16h;
typedef __attribute__((ext_vector_type(8)))  _Float16 v8h;
typedef __attribute__((ext_vector_type(8)))  float    v8f;
typedef __attribute__((ext_vector_type(4)))  unsigned u32x4;
typedef __attribute__((ext_vector_type(8)))  unsigned u32x8;

// ---------------- utility kernels ----------------
__global__ void gcn_zero(float* __restrict__ p, int n) {
  int i = blockIdx.x * blockDim.x + threadIdx.x;
  if (i < n) p[i] = 0.f;
}

__global__ void gcn_deg(const int* __restrict__ dst, float* __restrict__ deg, int ne) {
  int i = blockIdx.x * blockDim.x + threadIdx.x;
  if (i < ne) atomicAdd(&deg[dst[i]], 1.0f);
}

__global__ void gcn_dinv(float* __restrict__ deg, int n) {
  int i = blockIdx.x * blockDim.x + threadIdx.x;
  if (i < n) deg[i] = rsqrtf(deg[i] + 1.0f);   // +1 = self loop; always > 0
}

// h1[n,f] = sum_k x[n,k] * W1[k,f]
__global__ void gcn_xw1(const float* __restrict__ x, const float* __restrict__ W1,
                        float* __restrict__ h, int n) {
  int gid = blockIdx.x * blockDim.x + threadIdx.x;
  int node = gid >> 6, f = gid & 63;
  if (node >= n) return;
  float x0 = x[node*3+0], x1 = x[node*3+1], x2 = x[node*3+2];
  h[gid] = x0 * W1[f] + x1 * W1[64+f] + x2 * W1[128+f];
}

// hout = dinv[n]^2 * hin  (self-loop term initializes conv accumulator)
__global__ void gcn_self(const float* __restrict__ hin, const float* __restrict__ dinv,
                         float* __restrict__ hout, int n) {
  int gid = blockIdx.x * blockDim.x + threadIdx.x;
  int node = gid >> 6;
  if (node >= n) return;
  float di = dinv[node];
  hout[gid] = di * di * hin[gid];
}

// scatter: hout[dst] += dinv[src]*dinv[dst] * hin[src]; 16 lanes per edge, float4 each
__global__ void gcn_scatter(const int* __restrict__ src, const int* __restrict__ dstv,
                            const float* __restrict__ dinv,
                            const float* __restrict__ hin, float* __restrict__ hout, int ne) {
  int gid = blockIdx.x * blockDim.x + threadIdx.x;
  int e = gid >> 4, q = gid & 15;
  if (e >= ne) return;
  if (q == 0 && e + 4096 < ne) {           // stream-ahead: edge lists come from HBM
    __builtin_prefetch(src + e + 4096, 0, 0);
    __builtin_prefetch(dstv + e + 4096, 0, 0);
  }
  int s = src[e], d = dstv[e];
  float nrm = dinv[s] * dinv[d];
  float4 v = ((const float4*)(hin + (size_t)s * 64))[q];
  float* o = hout + (size_t)d * 64 + (size_t)q * 4;
  atomicAdd(o + 0, nrm * v.x);
  atomicAdd(o + 1, nrm * v.y);
  atomicAdd(o + 2, nrm * v.z);
  atomicAdd(o + 3, nrm * v.w);
}

__global__ void gcn_bias_relu(float* __restrict__ h, const float* __restrict__ b, int n) {
  int gid = blockIdx.x * blockDim.x + threadIdx.x;
  int node = gid >> 6, f = gid & 63;
  if (node < n) h[gid] = fmaxf(h[gid] + b[f], 0.f);
}

// ---------------- WMMA GEMM: out[N,64] = A[N,64] @ W[64,64] ----------------
// A tile staged to LDS with the Tensor Data Mover (tensor_load_to_lds, TENSORcnt),
// f32->f16 conversion done register-side while packing WMMA fragments.
__global__ void __launch_bounds__(256)
gcn_gemm64(const float* __restrict__ A, const float* __restrict__ W,
           float* __restrict__ out, int nrows) {
  __shared__ __align__(16)  _Float16 sBt[64 * 64];    // W transposed f16: [n][k]
  __shared__ __align__(128) float    sAf[8][16 * 64]; // per-wave 16x64 f32 tile (TDM dest)
  const int tid  = threadIdx.x;
  const int wave = tid >> 5;
  const int lane = tid & 31;

  // stage W2 into LDS transposed + converted to f16
  for (int i = tid; i < 64 * 64; i += 256) {
    int k = i >> 6, n = i & 63;
    sBt[n * 64 + k] = (_Float16)W[i];
  }
  __syncthreads();

  const int rowTile = blockIdx.x * 8 + wave;
  const int row0 = rowTile * 16;
  if (row0 >= nrows) return;
  const int rowsHere = (nrows - row0 < 16) ? (nrows - row0) : 16;

  float* myAf = &sAf[wave][0];

  // ---- TDM: DMA the 16x64 f32 tile from global into this wave's LDS slot ----
  {
    unsigned long long ga = (unsigned long long)(const void*)(A + (size_t)row0 * 64);
    unsigned lds_off = (unsigned)(unsigned long long)(const void*)myAf; // LDS aperture low bits
    u32x4 dg0;
    dg0[0] = 1u;                                   // count=1, user descriptor
    dg0[1] = lds_off;                              // lds_addr
    dg0[2] = (unsigned)ga;                         // global_addr[31:0]
    dg0[3] = (unsigned)((ga >> 32) & 0x01FFFFFFull) | 0x80000000u; // addr[56:32] | type=2
    u32x8 dg1;
    dg1[0] = 0x00020000u;                          // data_size=2 -> 4-byte elements
    dg1[1] = 64u << 16;                            // tensor_dim0 = 64 (row length)
    dg1[2] = ((unsigned)rowsHere) << 16;           // tensor_dim1 = rows (OOB rows -> 0)
    dg1[3] = 64u << 16;                            // tile_dim0 = 64
    dg1[4] = 16u;                                  // tile_dim1 = 16, tile_dim2 = 0
    dg1[5] = 64u;                                  // tensor_dim0_stride = 64 elements
    dg1[6] = 0u;
    dg1[7] = 0u;
    asm volatile("tensor_load_to_lds %0, %1" :: "s"(dg0), "s"(dg1) : "memory");
  }
  __builtin_amdgcn_s_wait_tensorcnt(0);

  const int half = lane >> 4;   // lane half (0/1)
  const int mn   = lane & 15;   // M for A rows, N for B/C columns

  v8f acc[4] = {v8f{}, v8f{}, v8f{}, v8f{}};
#pragma unroll
  for (int kstep = 0; kstep < 2; ++kstep) {
    const int kbase = kstep * 32;
    // A fragment (ISA 7.12.2 16-bit A 16x32 layout): two contiguous 8-elem runs,
    // loaded as f32 from LDS and converted to f16 in registers.
    const float* ab = myAf + mn * 64 + kbase + half * 8;
    v16h afrag;
#pragma unroll
    for (int j = 0; j < 8; ++j) {
      afrag[j]     = (_Float16)ab[j];
      afrag[8 + j] = (_Float16)ab[16 + j];
    }
#pragma unroll
    for (int nt = 0; nt < 4; ++nt) {
      // B fragment: same K order as A (runs at kbase+8*half and kbase+16+8*half)
      const _Float16* bb = sBt + (size_t)(nt * 16 + mn) * 64 + kbase + half * 8;
      v8h b0 = *(const v8h*)(bb);
      v8h b1 = *(const v8h*)(bb + 16);
      v16h bfrag;
#pragma unroll
      for (int j = 0; j < 8; ++j) { bfrag[j] = b0[j]; bfrag[8 + j] = b1[j]; }
      acc[nt] = __builtin_amdgcn_wmma_f32_16x16x32_f16(
          false, afrag, false, bfrag, (short)0, acc[nt], false, false);
    }
  }

  // C/D layout: VGPR p -> M = p + 8*half, N = mn + 16*nt
  if (rowsHere == 16) {          // full tile: branch-free store path
#pragma unroll
    for (int nt = 0; nt < 4; ++nt) {
#pragma unroll
      for (int p = 0; p < 8; ++p) {
        int gm = row0 + half * 8 + p;
        int gn = nt * 16 + mn;
        out[(size_t)gm * 64 + gn] = acc[nt][p];
      }
    }
  } else {
#pragma unroll
    for (int nt = 0; nt < 4; ++nt) {
#pragma unroll
      for (int p = 0; p < 8; ++p) {
        int gm = row0 + half * 8 + p;
        int gn = nt * 16 + mn;
        if (gm < nrows) out[(size_t)gm * 64 + gn] = acc[nt][p];
      }
    }
  }
}

// ---------------- pooling + readout ----------------
__global__ void gcn_pool(const float* __restrict__ h, const float* __restrict__ b2,
                         const int* __restrict__ batch, float* __restrict__ pooled,
                         float* __restrict__ counts, int n) {
  int gid = blockIdx.x * blockDim.x + threadIdx.x;
  int node = gid >> 6, f = gid & 63;
  if (node >= n) return;
  float v = fmaxf(h[gid] + b2[f], 0.f);
  int g = batch[node];
  atomicAdd(&pooled[g * 64 + f], v);
  if (f == 0) atomicAdd(&counts[g], 1.0f);
}

__global__ void gcn_readout(const float* __restrict__ pooled, const float* __restrict__ counts,
                            const float* __restrict__ Wr, const float* __restrict__ br,
                            float* __restrict__ out) {
  int g = threadIdx.x;
  if (g >= 64) return;
  float c = fmaxf(counts[g], 1.0f);
  float acc = 0.f;
  for (int f = 0; f < 64; ++f) acc += pooled[g * 64 + f] * Wr[f];
  out[g] = acc / c + br[0];
}

// ---------------- launcher ----------------
extern "C" void kernel_launch(void* const* d_in, const int* in_sizes, int n_in,
                              void* d_out, int out_size, void* d_ws, size_t ws_size,
                              hipStream_t stream) {
  const float* x     = (const float*)d_in[0];
  const int*   ei    = (const int*)d_in[1];
  const int*   batch = (const int*)d_in[2];
  const float* W1    = (const float*)d_in[3];
  const float* b1    = (const float*)d_in[4];
  const float* W2    = (const float*)d_in[5];
  const float* b2    = (const float*)d_in[6];
  const float* Wr    = (const float*)d_in[7];
  const float* br    = (const float*)d_in[8];
  float* out = (float*)d_out;

  const int N = in_sizes[0] / 3;
  const int E = in_sizes[1] / 2;
  const int* srcIdx = ei;        // edge_index[0]
  const int* dstIdx = ei + E;    // edge_index[1]

  float* ws     = (float*)d_ws;
  float* dinv   = ws;                                      // N
  float* hP1    = ws + (((size_t)N + 255) & ~(size_t)255); // N*64
  float* hP2    = hP1 + (size_t)N * 64;                    // N*64
  float* pooled = hP2 + (size_t)N * 64;                    // 64*64
  float* counts = pooled + 64 * 64;                        // 64 (contiguous after pooled)

  const int NF = N * 64;
  dim3 B(256);

  // degrees -> dinv
  gcn_zero<<<(N + 255) / 256, B, 0, stream>>>(dinv, N);
  gcn_deg<<<(E + 255) / 256, B, 0, stream>>>(dstIdx, dinv, E);
  gcn_dinv<<<(N + 255) / 256, B, 0, stream>>>(dinv, N);

  // layer 1: hP1 = x@W1 ; hP2 = conv(hP1) ; relu(+b1)
  gcn_xw1<<<(NF + 255) / 256, B, 0, stream>>>(x, W1, hP1, N);
  gcn_self<<<(NF + 255) / 256, B, 0, stream>>>(hP1, dinv, hP2, N);
  gcn_scatter<<<(E * 16 + 255) / 256, B, 0, stream>>>(srcIdx, dstIdx, dinv, hP1, hP2, E);
  gcn_bias_relu<<<(NF + 255) / 256, B, 0, stream>>>(hP2, b1, N);

  // layer 2: hP1 = hP2 @ W2 (WMMA + TDM), then hP2 = conv(hP1)
  int tiles = (N + 15) / 16;
  gcn_gemm64<<<(tiles + 7) / 8, B, 0, stream>>>(hP2, W2, hP1, N);
  gcn_self<<<(NF + 255) / 256, B, 0, stream>>>(hP1, dinv, hP2, N);
  gcn_scatter<<<(E * 16 + 255) / 256, B, 0, stream>>>(srcIdx, dstIdx, dinv, hP1, hP2, E);

  // pooling (+b2, relu fused) and readout
  gcn_zero<<<(64 * 65 + 255) / 256, B, 0, stream>>>(pooled, 64 * 65);
  gcn_pool<<<(NF + 255) / 256, B, 0, stream>>>(hP2, b2, batch, pooled, counts, N);
  gcn_readout<<<1, 64, 0, stream>>>(pooled, counts, Wr, br, out);
}
